// RandomWalkPositionalEncoding_56573309223852
// MI455X (gfx1250) — matrix operand order repaired and checked by reading, not compiled, not run on hardware
//
#include <hip/hip_runtime.h>
#include <hip/hip_bf16.h>
#include <stdint.h>

typedef __attribute__((ext_vector_type(16))) __bf16 v16bf;
typedef __attribute__((ext_vector_type(8)))  float  v8f;

#define N      8192
#define STEPS  10
#define EMBED  32
#define NEDGE  131072

#define BM 128
#define BN 128
#define BK 32
#define LDA 40   // padded LDS row stride in ushorts (80B, multiple of 16B)
#define KTILES (N / BK)

// ---------------- CDNA5 async copy helpers ----------------

__device__ __forceinline__ unsigned lds_off(const void* p) {
    // generic pointer to LDS: low 32 bits are the LDS byte offset (aperture mapping)
    return (unsigned)(unsigned long long)p;
}

__device__ __forceinline__ void async_b128(unsigned lds, const void* g) {
    // GLOBAL_LOAD_ASYNC_TO_LDS_B128: per-lane 16B global -> LDS, tracked by ASYNCcnt
    asm volatile("global_load_async_to_lds_b128 %0, %1, off"
                 :: "v"(lds), "v"((unsigned long long)g)
                 : "memory");
}

__device__ __forceinline__ void wait_async0() {
    asm volatile("s_wait_asynccnt 0x0" ::: "memory");
}

// ---------------- utility kernels ----------------

__global__ void zero_u32_kernel(uint32_t* p, size_t n) {
    size_t i = (size_t)blockIdx.x * blockDim.x + threadIdx.x;
    size_t stride = (size_t)gridDim.x * blockDim.x;
    for (; i < n; i += stride) p[i] = 0u;
}

__global__ void scatter_edges_kernel(const long long* __restrict__ e,
                                     unsigned short* __restrict__ adj) {
    int i = blockIdx.x * blockDim.x + threadIdx.x;
    if (i >= NEDGE) return;
    int r = (int)e[i];
    int c = (int)e[NEDGE + i];
    adj[(size_t)r * N + c] = 0x3F80u;   // bf16(1.0)
    adj[(size_t)c * N + r] = 0x3F80u;   // symmetric, set semantics
}

// one block (256 threads) per row: count nonzeros, write bf16(1/(deg+eps)) in place
__global__ void row_normalize_kernel(unsigned short* __restrict__ adj) {
    __shared__ float red[256];
    const int row = blockIdx.x;
    const size_t base = (size_t)row * N;
    int cnt = 0;
    for (int j = threadIdx.x; j < N; j += 256)
        cnt += (adj[base + j] != 0);
    red[threadIdx.x] = (float)cnt;
    __syncthreads();
    for (int s = 128; s > 0; s >>= 1) {
        if (threadIdx.x < (unsigned)s) red[threadIdx.x] += red[threadIdx.x + s];
        __syncthreads();
    }
    const float inv = 1.0f / (red[0] + 1e-8f);
    union { __bf16 b; unsigned short u; } cv;
    cv.b = (__bf16)inv;
    for (int j = threadIdx.x; j < N; j += 256)
        if (adj[base + j] != 0) adj[base + j] = cv.u;
}

__global__ void diag_step0_kernel(const __bf16* __restrict__ adjn,
                                  float* __restrict__ out) {
    int i = blockIdx.x * blockDim.x + threadIdx.x;
    if (i < N) out[(size_t)i * EMBED + 0] = (float)adjn[(size_t)i * N + i];
}

// ---------------- WMMA GEMM: C = A @ B (bf16 in, fp32 acc, bf16 out) ----------------
// Double-buffered LDS; A staged via global_load_async_to_lds_b128 (ASYNCcnt),
// B staged via VGPR + transposed b16 stores (needs transpose into [n][k]).
// Writes fp32 diagonal entries into out[row*EMBED + step].

__global__ __launch_bounds__(256)
void gemm_diag_kernel(const __bf16* __restrict__ A,
                      const __bf16* __restrict__ B,
                      __bf16* __restrict__ C,
                      float* __restrict__ out,
                      int step) {
    __shared__ alignas(16) unsigned short As[2][BM * LDA];   // row-major [m][k]
    __shared__ alignas(16) unsigned short Bs[2][BN * LDA];   // transposed [n][k]

    const int tid  = threadIdx.x;
    const int lane = tid & 31;
    const int wave = tid >> 5;      // 8 waves
    const int wm   = wave & 1;      // 2 wave rows  -> 64 rows each
    const int wn   = wave >> 1;     // 4 wave cols  -> 32 cols each
    const int bx   = blockIdx.x;    // N tile
    const int by   = blockIdx.y;    // M tile

    v8f acc[4][2];
#pragma unroll
    for (int i = 0; i < 4; ++i)
#pragma unroll
        for (int j = 0; j < 2; ++j)
            acc[i][j] = (v8f){0.f, 0.f, 0.f, 0.f, 0.f, 0.f, 0.f, 0.f};

    // staging indices: each thread moves 16 bf16 (2 x b128) for A and B
    const int arow = tid >> 1;            // 0..127
    const int acol = (tid & 1) << 4;      // 0 or 16 (ushorts)
    const int brow = tid >> 3;            // 0..31 (k)
    const int bcol = (tid & 7) << 4;      // 0..112 (n)

    const __bf16* Ap = A + (size_t)(by * BM + arow) * N + acol;
    const __bf16* Bp = B + (size_t)brow * N + (size_t)bx * BN + bcol;

    const unsigned aLds0 = lds_off(&As[0][arow * LDA + acol]);
    const unsigned aLds1 = lds_off(&As[1][arow * LDA + acol]);

    const int mrow   = lane & 15;
    const int khalfA = (lane < 16) ? 0 : 8;    // A: lanes 16-31 hold K 8..15 / 24..31
    const int khalfB = (lane < 16) ? 0 : 8;    // B: lanes 16-31 hold K 8..15 / 24..31

    // ---- prologue: stage tile 0 into buffer 0 ----
    {
        async_b128(aLds0,      Ap);
        async_b128(aLds0 + 16, Ap + 8);
        const uint4* bg = (const uint4*)Bp;
        uint4 b0 = bg[0];
        uint4 b1 = bg[1];
        unsigned short tmp[16];
        *(uint4*)&tmp[0] = b0;
        *(uint4*)&tmp[8] = b1;
#pragma unroll
        for (int e = 0; e < 16; ++e)
            Bs[0][(bcol + e) * LDA + brow] = tmp[e];
        wait_async0();
    }
    __syncthreads();   // tile 0 resident

    for (int kt = 0; kt < KTILES; ++kt) {
        const int cur  = kt & 1;
        const int nxt  = cur ^ 1;
        const bool more = (kt + 1 < KTILES);

        uint4 nb0, nb1;
        if (more) {
            // async A copy of next tile directly into the other LDS buffer
            const __bf16* ga = Ap + (size_t)(kt + 1) * BK;
            async_b128(nxt ? aLds1 : aLds0, ga);
            async_b128((nxt ? aLds1 : aLds0) + 16, ga + 8);
            // B next tile into VGPRs (transpose on LDS store later)
            const uint4* bg = (const uint4*)(Bp + (size_t)(kt + 1) * BK * N);
            nb0 = bg[0];
            nb1 = bg[1];
        }
        if (kt + 2 < KTILES)
            __builtin_prefetch(Bp + (size_t)(kt + 2) * BK * N, 0, 0);

        // ---- compute tile kt from buffer cur ----
        v16bf af[4], bfr[2];
#pragma unroll
        for (int i = 0; i < 4; ++i) {
            const unsigned short* p = &As[cur][(wm * 64 + i * 16 + mrow) * LDA + khalfA];
            ((uint4*)&af[i])[0] = *(const uint4*)(p);        // K 0..7   (or 8..15)
            ((uint4*)&af[i])[1] = *(const uint4*)(p + 16);   // K 16..23 (or 24..31)
        }
#pragma unroll
        for (int j = 0; j < 2; ++j) {
            const unsigned short* p = &Bs[cur][(wn * 32 + j * 16 + mrow) * LDA + khalfB];
            ((uint4*)&bfr[j])[0] = *(const uint4*)(p);       // K 0..7   (or 8..15)
            ((uint4*)&bfr[j])[1] = *(const uint4*)(p + 16);  // K 16..23 (or 24..31)
        }
#pragma unroll
        for (int i = 0; i < 4; ++i)
#pragma unroll
            for (int j = 0; j < 2; ++j)
                acc[i][j] = __builtin_amdgcn_wmma_f32_16x16x32_bf16(
                    false, af[i], false, bfr[j], (short)0, acc[i][j], false, false);

        if (more) {
            // transpose-store next B tile; buffer nxt's last readers finished
            // before the barrier we passed entering this iteration
            unsigned short tmp[16];
            *(uint4*)&tmp[0] = nb0;
            *(uint4*)&tmp[8] = nb1;
#pragma unroll
            for (int e = 0; e < 16; ++e)
                Bs[nxt][(bcol + e) * LDA + brow] = tmp[e];
            wait_async0();      // own A asyncs for tile kt+1 landed
            __syncthreads();    // everyone's tile kt+1 resident; tile kt reads retired
        }
    }

    // ---- epilogue: bf16 store of C, fp32 diagonal capture ----
    const int cn = lane & 15;
    const int rh = (lane >> 4) << 3;   // 0 or 8
#pragma unroll
    for (int i = 0; i < 4; ++i) {
#pragma unroll
        for (int j = 0; j < 2; ++j) {
            const int gm0 = by * BM + wm * 64 + i * 16;
            const int gn  = bx * BN + wn * 32 + j * 16 + cn;
#pragma unroll
            for (int e = 0; e < 8; ++e) {
                const int gr  = gm0 + rh + e;
                const float v = acc[i][j][e];
                C[(size_t)gr * N + gn] = (__bf16)v;
                if (gr == gn)
                    out[(size_t)gr * EMBED + step] = v;
            }
        }
    }
}

// ---------------- launcher ----------------

extern "C" void kernel_launch(void* const* d_in, const int* in_sizes, int n_in,
                              void* d_out, int out_size, void* d_ws, size_t ws_size,
                              hipStream_t stream) {
    (void)in_sizes; (void)n_in; (void)ws_size;
    const long long* edges = (const long long*)d_in[1];   // d_in[0] = x (unused)
    float* out = (float*)d_out;

    char* ws = (char*)d_ws;
    const size_t matB = (size_t)N * N * sizeof(unsigned short);  // 128 MB
    unsigned short* adj_us = (unsigned short*)ws;
    __bf16* adjn = (__bf16*)ws;
    __bf16* c0   = (__bf16*)(ws + matB);
    __bf16* c1   = (__bf16*)(ws + 2 * matB);

    // zero adjacency workspace and output
    zero_u32_kernel<<<4096, 256, 0, stream>>>((uint32_t*)ws, matB / 4);
    zero_u32_kernel<<<256, 256, 0, stream>>>((uint32_t*)d_out, (size_t)out_size);

    scatter_edges_kernel<<<NEDGE / 256, 256, 0, stream>>>(edges, adj_us);
    row_normalize_kernel<<<N, 256, 0, stream>>>(adj_us);
    diag_step0_kernel<<<N / 256, 256, 0, stream>>>(adjn, out);

    // power chain: C_{k+1} = C_k @ M, M stationary (fits in 192MB L2)
    dim3 grid(N / BN, N / BM);
    const __bf16* curA = adjn;
    __bf16* dst = c0;
    for (int s = 1; s < STEPS; ++s) {
        gemm_diag_kernel<<<grid, 256, 0, stream>>>(curA, adjn, dst, out, s);
        curA = dst;
        dst  = (dst == c0) ? c1 : c0;
    }
}
